// GraphAttention_4587025072209
// MI455X (gfx1250) — matrix operand order repaired and checked by reading, not compiled, not run on hardware
//
#include <hip/hip_runtime.h>
#include <hip/hip_bf16.h>

typedef __attribute__((ext_vector_type(16))) _Float16 v16h;
typedef __attribute__((ext_vector_type(8)))  _Float16 v8h;
typedef __attribute__((ext_vector_type(8)))  float    v8f;

#define IN_DIM 128
#define NHEAD  8
#define HDIM   16
#define QKVW   128           // H*D
#define TOTCOL 384           // 3 * 128 concatenated Q|K|V columns

// ---------------------------------------------------------------------------
// Prep: fp32 -> f16 conversion of node features
// ---------------------------------------------------------------------------
__global__ __launch_bounds__(256) void ga_h_to_f16(const float* __restrict__ h,
                                                   _Float16* __restrict__ h16,
                                                   long total) {
  long i = (long)blockIdx.x * 256 + threadIdx.x;
  if (i < total) h16[i] = (_Float16)h[i];
}

// Prep: build transposed f16 weight matrix Wt[col][k], col in [0,384)
__global__ __launch_bounds__(256) void ga_w_transpose(const float* __restrict__ WQ,
                                                      const float* __restrict__ WK,
                                                      const float* __restrict__ WV,
                                                      _Float16* __restrict__ wt) {
  int i = blockIdx.x * 256 + threadIdx.x;          // 384*128 = 49152 total
  if (i >= TOTCOL * IN_DIM) return;
  int col = i >> 7;          // 0..383
  int k   = i & 127;         // 0..127
  int sel = col >> 7;        // 0:Q 1:K 2:V
  int c   = col & 127;
  const float* W = (sel == 0) ? WQ : (sel == 1) ? WK : WV;
  wt[i] = (_Float16)W[(size_t)k * QKVW + c];
}

__global__ __launch_bounds__(256) void ga_zero(float* __restrict__ p, long total) {
  long i = (long)blockIdx.x * 256 + threadIdx.x;
  if (i < total) p[i] = 0.0f;
}

// ---------------------------------------------------------------------------
// Phase 1: fused QKV projection via V_WMMA_F32_16X16X32_F16
//   block = 256 threads = 8 waves; each block computes a 16-row node tile
//   across all 384 output columns (24 16x16 tiles; 3 tiles per wave).
// ---------------------------------------------------------------------------
__global__ __launch_bounds__(256) void ga_qkv_wmma(const _Float16* __restrict__ h16,
                                                   const _Float16* __restrict__ wt16,
                                                   const float* __restrict__ bQ,
                                                   const float* __restrict__ bK,
                                                   const float* __restrict__ bV,
                                                   float* __restrict__ Q,
                                                   float* __restrict__ K,
                                                   float* __restrict__ V,
                                                   int N) {
  const int wave = threadIdx.x >> 5;
  const int lane = threadIdx.x & 31;
  const int hi   = lane >> 4;                 // lane half: 0 or 1
  const int r0   = blockIdx.x * 16;
  int arow_idx = r0 + (lane & 15);
  if (arow_idx >= N) arow_idx = N - 1;        // clamp (stores are guarded)
  const _Float16* __restrict__ arow = h16 + (size_t)arow_idx * IN_DIM;

#pragma unroll
  for (int t = 0; t < 3; ++t) {
    const int ct      = wave + t * 8;         // 0..23
    const int colbase = ct * 16;
    const int gcol    = colbase + (lane & 15);
    const _Float16* __restrict__ brow = wt16 + (size_t)gcol * IN_DIM;

    v8f acc = {};
#pragma unroll
    for (int k0 = 0; k0 < IN_DIM; k0 += 32) {
      // A fragment (16x32 f16): lane half 0 -> K k0+{0..15}; half 1 -> k0+{16..31}
      const int kb = k0 + (hi << 4);
      v8h alo = *(const v8h*)(arow + kb);
      v8h ahi = *(const v8h*)(arow + kb + 8);
      v16h a = __builtin_shufflevector(alo, ahi, 0, 1, 2, 3, 4, 5, 6, 7,
                                       8, 9, 10, 11, 12, 13, 14, 15);
      // B fragment (32x16 f16): lane = column, 16 contiguous K values
      const int ks = k0 + (hi << 4);
      v8h blo = *(const v8h*)(brow + ks);
      v8h bhi = *(const v8h*)(brow + ks + 8);
      v16h b = __builtin_shufflevector(blo, bhi, 0, 1, 2, 3, 4, 5, 6, 7,
                                       8, 9, 10, 11, 12, 13, 14, 15);
      acc = __builtin_amdgcn_wmma_f32_16x16x32_f16(
          /*neg_a=*/false, a, /*neg_b=*/false, b,
          /*c_mod=*/(short)0, acc, /*reuse_a=*/false, /*reuse_b=*/false);
    }

    // bias + store (C/D layout: vgpr v -> row v + 8*hi, col = lane&15)
    const int sel = gcol >> 7;
    const int c   = gcol & 127;
    const float bias = (sel == 0) ? bQ[c] : (sel == 1) ? bK[c] : bV[c];
    float* __restrict__ outp = (sel == 0) ? Q : (sel == 1) ? K : V;
#pragma unroll
    for (int v = 0; v < 8; ++v) {
      const int m = v + (hi << 3);
      const int row = r0 + m;
      if (row < N) outp[(size_t)row * QKVW + c] = acc[v] + bias;
    }
  }
}

// ---------------------------------------------------------------------------
// Phase 2: edge scores + scatter. One wave32 per edge.
//   lane l owns features [4l,4l+4); head = l>>2 reduced over its 4 lanes.
// ---------------------------------------------------------------------------
__global__ __launch_bounds__(256) void ga_edge(const int* __restrict__ src,
                                               const int* __restrict__ dst,
                                               const float* __restrict__ Q,
                                               const float* __restrict__ K,
                                               const float* __restrict__ V,
                                               float* __restrict__ out,
                                               float* __restrict__ z,
                                               int E) {
  const int wave = threadIdx.x >> 5;
  const int lane = threadIdx.x & 31;
  const int e = blockIdx.x * 8 + wave;
  if (e >= E) return;

  const int s = src[e];
  const int d = dst[e];
  const int j = lane << 2;                    // feature offset 0..124

  const float4 kv = *(const float4*)(K + (size_t)s * QKVW + j);
  const float4 qv = *(const float4*)(Q + (size_t)d * QKVW + j);
  float p = kv.x * qv.x + kv.y * qv.y + kv.z * qv.z + kv.w * qv.w;
  // butterfly reduce within the 4-lane head group (wave32 shuffles)
  p += __shfl_xor(p, 1, 32);
  p += __shfl_xor(p, 2, 32);

  float sc = p * 0.25f;                       // / sqrt(D=16)
  sc = fminf(5.0f, fmaxf(-5.0f, sc));
  const float score = __expf(sc);

  const float4 vv = *(const float4*)(V + (size_t)s * QKVW + j);
  float* __restrict__ o = out + (size_t)d * QKVW + j;
  atomicAdd(o + 0, score * vv.x);
  atomicAdd(o + 1, score * vv.y);
  atomicAdd(o + 2, score * vv.z);
  atomicAdd(o + 3, score * vv.w);
  if ((lane & 3) == 0) atomicAdd(z + (size_t)d * NHEAD + (lane >> 2), score);
}

// ---------------------------------------------------------------------------
// Phase 3: normalize in place: out[n,h,d] /= z[n,h]
// ---------------------------------------------------------------------------
__global__ __launch_bounds__(256) void ga_norm(float* __restrict__ out,
                                               const float* __restrict__ z,
                                               long total) {
  long i = (long)blockIdx.x * 256 + threadIdx.x;
  if (i < total) {
    const long n  = i >> 7;
    const int hh  = (int)((i >> 4) & 7);
    out[i] = out[i] / z[n * NHEAD + hh];
  }
}

// ---------------------------------------------------------------------------
extern "C" void kernel_launch(void* const* d_in, const int* in_sizes, int n_in,
                              void* d_out, int out_size, void* d_ws, size_t ws_size,
                              hipStream_t stream) {
  const float* h   = (const float*)d_in[0];
  const int*   src = (const int*)d_in[1];
  const int*   dst = (const int*)d_in[2];
  const float* WQ  = (const float*)d_in[3];
  const float* WK  = (const float*)d_in[4];
  const float* WV  = (const float*)d_in[5];
  const float* bQ  = (const float*)d_in[6];
  const float* bK  = (const float*)d_in[7];
  const float* bV  = (const float*)d_in[8];
  float* out = (float*)d_out;

  const int N = in_sizes[0] / IN_DIM;
  const int E = in_sizes[1];

  // workspace carve-up (256B aligned slabs)
  char* ws = (char*)d_ws;
  size_t off = 0;
  _Float16* h16 = (_Float16*)(ws + off); off += (size_t)N * IN_DIM * sizeof(_Float16);
  off = (off + 255) & ~(size_t)255;
  _Float16* wt16 = (_Float16*)(ws + off); off += (size_t)TOTCOL * IN_DIM * sizeof(_Float16);
  off = (off + 255) & ~(size_t)255;
  float* Q = (float*)(ws + off); off += (size_t)N * QKVW * sizeof(float);
  float* K = (float*)(ws + off); off += (size_t)N * QKVW * sizeof(float);
  float* V = (float*)(ws + off); off += (size_t)N * QKVW * sizeof(float);
  float* z = (float*)(ws + off); off += (size_t)N * NHEAD * sizeof(float);

  const long hTot = (long)N * IN_DIM;          // also = out element count
  const long zTot = (long)N * NHEAD;

  // prep
  ga_h_to_f16<<<dim3((unsigned)((hTot + 255) / 256)), dim3(256), 0, stream>>>(h, h16, hTot);
  ga_w_transpose<<<dim3((TOTCOL * IN_DIM + 255) / 256), dim3(256), 0, stream>>>(WQ, WK, WV, wt16);
  ga_zero<<<dim3((unsigned)((hTot + 255) / 256)), dim3(256), 0, stream>>>(out, hTot);
  ga_zero<<<dim3((unsigned)((zTot + 255) / 256)), dim3(256), 0, stream>>>(z, zTot);

  // phase 1: WMMA QKV projection (16 node-rows per block)
  ga_qkv_wmma<<<dim3((N + 15) / 16), dim3(256), 0, stream>>>(h16, wt16, bQ, bK, bV, Q, K, V, N);

  // phase 2: per-edge attention scatter (8 edges per 256-thread block)
  ga_edge<<<dim3((E + 7) / 8), dim3(256), 0, stream>>>(src, dst, Q, K, V, out, z, E);

  // phase 3: normalize
  ga_norm<<<dim3((unsigned)((hTot + 255) / 256)), dim3(256), 0, stream>>>(out, z, hTot);
}
